// ProteinFeatures_20779051778384
// MI455X (gfx1250) — compile-verified, matrix operand order checked by reading, not run on hardware
//
#include <hip/hip_runtime.h>
#include <hip/hip_bf16.h>

typedef __attribute__((ext_vector_type(8)))  float  v8f;
typedef __attribute__((ext_vector_type(16))) __bf16 v16bf;
typedef __attribute__((ext_vector_type(8)))  __bf16 v8bf;

#define NNODES   1024
#define TOPK     30
#define NEDGES   (NNODES * TOPK)          // 30720
#define NATOMS   14
#define NPAIR    (NATOMS * NATOMS)        // 196
#define NRBF     16
#define NPE      16
#define KREAL    (NPE + NPAIR * NRBF)     // 3152
#define KPAD     3200                     // 100 chunks of K=32
#define NKC      (KPAD / 32)              // 100
#define DMODEL   128

// workspace layout (bytes)
#define WS_XA    0                        // 1024*42 f32  = 172032 B
#define WS_EIDX  172032                   // 30720 i32    = 122880 B
#define WS_WPK   294912                   // 100*32*128 bf16 = 819200 B

// output layout (f32 elements)
#define OUT_V    0
#define OUT_E    (NNODES * DMODEL)                    // 131072
#define OUT_EIDX (OUT_E + NEDGES * DMODEL)            // 4063232

// ---------------------------------------------------------------- Xa build
__global__ __launch_bounds__(256) void xa_kernel(const float* __restrict__ X,
                                                 float* __restrict__ Xa) {
  int i = blockIdx.x * 256 + threadIdx.x;
  if (i >= NNODES) return;
  const float* xi = X + (size_t)i * NATOMS * 3;
  float Nn[3], Ca[3], Cc[3], Oo[3], b[3], c[3], a[3], Cb[3];
#pragma unroll
  for (int d = 0; d < 3; ++d) {
    Nn[d] = xi[0 * 3 + d]; Ca[d] = xi[1 * 3 + d];
    Cc[d] = xi[2 * 3 + d]; Oo[d] = xi[3 * 3 + d];
    b[d] = Ca[d] - Nn[d];  c[d] = Cc[d] - Ca[d];
  }
  a[0] = b[1] * c[2] - b[2] * c[1];
  a[1] = b[2] * c[0] - b[0] * c[2];
  a[2] = b[0] * c[1] - b[1] * c[0];
#pragma unroll
  for (int d = 0; d < 3; ++d)
    Cb[d] = -0.58273431f * a[d] + 0.56802827f * b[d] - 0.54067466f * c[d] + Ca[d];
  float* o = Xa + (size_t)i * NATOMS * 3;
#pragma unroll
  for (int d = 0; d < 3; ++d) {
    o[0 * 3 + d] = Nn[d]; o[1 * 3 + d] = Ca[d]; o[2 * 3 + d] = Cc[d];
    o[3 * 3 + d] = Oo[d]; o[4 * 3 + d] = Cb[d];
  }
  for (int r = 5; r < NATOMS; ++r)
#pragma unroll
    for (int d = 0; d < 3; ++d) o[r * 3 + d] = xi[r * 3 + d];
}

// ---------------------------------------------------------------- KNN
__global__ __launch_bounds__(256) void knn_kernel(const float* __restrict__ X,
                                                  const float* __restrict__ mask,
                                                  int* __restrict__ Eidx,
                                                  float* __restrict__ EidxF) {
  __shared__ float Dls[NNODES];
  __shared__ float redV[256];
  __shared__ int   redI[256];
  const int i = blockIdx.x;
  const int t = threadIdx.x;
  const float mi = mask[i];
  const float cx = X[((size_t)i * NATOMS + 1) * 3 + 0];
  const float cy = X[((size_t)i * NATOMS + 1) * 3 + 1];
  const float cz = X[((size_t)i * NATOMS + 1) * 3 + 2];
  for (int j = t; j < NNODES; j += 256) {
    float dx = cx - X[((size_t)j * NATOMS + 1) * 3 + 0];
    float dy = cy - X[((size_t)j * NATOMS + 1) * 3 + 1];
    float dz = cz - X[((size_t)j * NATOMS + 1) * 3 + 2];
    float m2 = mi * mask[j];
    Dls[j] = m2 * sqrtf(dx * dx + dy * dy + dz * dz + 1e-6f);
  }
  __syncthreads();
  // D_max over row
  float mx = -3.0e38f;
  for (int j = t; j < NNODES; j += 256) mx = fmaxf(mx, Dls[j]);
  redV[t] = mx;
  __syncthreads();
  for (int s = 128; s > 0; s >>= 1) {
    if (t < s) redV[t] = fmaxf(redV[t], redV[t + s]);
    __syncthreads();
  }
  const float Dmax = redV[0];
  // adjust: D + 2*(1-m2)*Dmax
  for (int j = t; j < NNODES; j += 256) {
    float m2 = mi * mask[j];
    Dls[j] += 2.0f * (1.0f - m2) * Dmax;
  }
  __syncthreads();
  // 30 argmin extractions, ties -> lowest index (jax top_k semantics)
  for (int kk = 0; kk < TOPK; ++kk) {
    float bv = 3.0e38f; int bi = 0x3fffffff;
    for (int j = t; j < NNODES; j += 256) {
      float v = Dls[j];
      if (v < bv || (v == bv && j < bi)) { bv = v; bi = j; }
    }
    redV[t] = bv; redI[t] = bi;
    __syncthreads();
    for (int s = 128; s > 0; s >>= 1) {
      if (t < s) {
        float v = redV[t + s]; int jj = redI[t + s];
        if (v < redV[t] || (v == redV[t] && jj < redI[t])) { redV[t] = v; redI[t] = jj; }
      }
      __syncthreads();
    }
    if (t == 0) {
      int j = redI[0];
      Eidx[i * TOPK + kk]  = j;
      EidxF[i * TOPK + kk] = (float)j;
      Dls[j] = 3.0e38f;
    }
    __syncthreads();
  }
}

// ---------------------------------------------------------------- weight pack f32 -> bf16 (WMMA B chunk layout)
__global__ __launch_bounds__(256) void pack_kernel(const float* __restrict__ W,
                                                   __bf16* __restrict__ Wpack) {
  int gid = blockIdx.x * 256 + threadIdx.x;
  if (gid >= NKC * DMODEL * 32) return;
  int k  = gid & 31;
  int n  = (gid >> 5) & 127;
  int kc = gid >> 12;
  int K  = kc * 32 + k;
  float v = (K < KREAL) ? W[(size_t)K * DMODEL + n] : 0.0f;
  Wpack[gid] = (__bf16)v;   // Wpack[(kc*128+n)*32+k]
}

// ---------------------------------------------------------------- node embed + LN (one wave per node)
__global__ __launch_bounds__(256) void node_kernel(const int* __restrict__ S,
                                                   const float* __restrict__ BBD,
                                                   const float* __restrict__ W,
                                                   const float* __restrict__ bias,
                                                   const float* __restrict__ g,
                                                   const float* __restrict__ be,
                                                   float* __restrict__ outV) {
  const int wave = threadIdx.x >> 5, lane = threadIdx.x & 31;
  const int i = blockIdx.x * 8 + wave;
  const int s = S[i];
  float bb[6];
#pragma unroll
  for (int d = 0; d < 6; ++d) bb[d] = BBD[(size_t)i * 6 + d];
  float acc[4], sum = 0.0f, ss = 0.0f;
#pragma unroll
  for (int q = 0; q < 4; ++q) {
    int n = lane + 32 * q;
    float a = W[(size_t)s * DMODEL + n] + bias[n];
#pragma unroll
    for (int d = 0; d < 6; ++d) a += bb[d] * W[(size_t)(21 + d) * DMODEL + n];
    acc[q] = a; sum += a; ss += a * a;
  }
#pragma unroll
  for (int off = 16; off > 0; off >>= 1) {
    sum += __shfl_xor(sum, off, 32);
    ss  += __shfl_xor(ss,  off, 32);
  }
  float mean = sum * (1.0f / 128.0f);
  float var  = ss * (1.0f / 128.0f) - mean * mean;
  float rstd = rsqrtf(var + 1e-5f);
#pragma unroll
  for (int q = 0; q < 4; ++q) {
    int n = lane + 32 * q;
    outV[(size_t)i * DMODEL + n] = (acc[q] - mean) * rstd * g[n] + be[n];
  }
}

// ---------------------------------------------------------------- fused edge features + bf16 WMMA GEMM + LN
// Full 16x3200 bf16 A-tile built once in LDS (CDNA5: up to 320KB LDS/WG),
// then a barrier-free 100-iteration WMMA K-loop.
__global__ __launch_bounds__(256) void edge_kernel(const float* __restrict__ Xa,
                                                   const int* __restrict__ Eidx,
                                                   const __bf16* __restrict__ Wpack,
                                                   const float* __restrict__ bias,
                                                   const float* __restrict__ g,
                                                   const float* __restrict__ be,
                                                   float* __restrict__ outE) {
  __shared__ float XaI[16][NATOMS * 3];
  __shared__ float XaJ[16][NATOMS * 3];
  __shared__ __attribute__((aligned(16))) __bf16 Abuf[16][KPAD];   // 102400 B
  __shared__ float Etile[16][DMODEL];
  __shared__ int   eJ[16];

  const int t = threadIdx.x;
  const int ebase = blockIdx.x * 16;

  if (t < 16) eJ[t] = Eidx[ebase + t];
  __syncthreads();

  // stage Xa tiles for the 16 edges of this block
  for (int idx = t; idx < 16 * NATOMS * 3; idx += 256) {
    int m = idx / (NATOMS * 3), c = idx % (NATOMS * 3);
    int i = (ebase + m) / TOPK;
    XaI[m][c] = Xa[(size_t)i * NATOMS * 3 + c];
    XaJ[m][c] = Xa[(size_t)eJ[m] * NATOMS * 3 + c];
  }
  __syncthreads();

  const float invsig = 16.0f / 20.0f;        // 1/sigma, sigma = 20/16
  const float mustep = 20.0f / 15.0f;        // linspace(0,20,16) step

  // positional encoding: 16 feats/edge, one per thread
  {
    int m = t >> 4, f = t & 15;
    int i = (ebase + m) / TOPK;
    float d   = (float)(eJ[m] - i);
    float fr  = __expf(-(float)((f & 7) * 2) * (9.210340371976184f / 16.0f));
    float ang = d * fr;
    Abuf[m][f] = (__bf16)((f < 8) ? __cosf(ang) : __sinf(ang));
  }
  // zero K padding [KREAL, KPAD)
  for (int idx = t; idx < 16 * (KPAD - KREAL); idx += 256) {
    int m = idx / (KPAD - KREAL), f = KREAL + idx % (KPAD - KREAL);
    Abuf[m][f] = (__bf16)0.0f;
  }
  // RBF: one task = one (edge, atom-pair): dist + 16 exps -> two 16B LDS stores
  for (int idx = t; idx < 16 * NPAIR; idx += 256) {
    int m = idx / NPAIR, p = idx % NPAIR;
    int a = p / NATOMS, b = p % NATOMS;
    float dx = XaI[m][a * 3 + 0] - XaJ[m][b * 3 + 0];
    float dy = XaI[m][a * 3 + 1] - XaJ[m][b * 3 + 1];
    float dz = XaI[m][a * 3 + 2] - XaJ[m][b * 3 + 2];
    float dist = sqrtf(dx * dx + dy * dy + dz * dz + 1e-6f);
    v8bf lo, hi;
#pragma unroll
    for (int mm = 0; mm < 8; ++mm) {
      float u0 = (dist - mustep * (float)mm) * invsig;
      float u1 = (dist - mustep * (float)(mm + 8)) * invsig;
      lo[mm] = (__bf16)__expf(-u0 * u0);
      hi[mm] = (__bf16)__expf(-u1 * u1);
    }
    *(v8bf*)&Abuf[m][NPE + p * 16]     = lo;
    *(v8bf*)&Abuf[m][NPE + p * 16 + 8] = hi;
  }
  __syncthreads();

  // ---------------- barrier-free WMMA K-loop ----------------
  const int wave  = t >> 5, lane = t & 31;
  const int nlo   = lane & 15;
  const int n     = wave * 16 + nlo;         // output column owned by this lane
  const int mrow  = lane & 15;               // A row for this lane
  const int khalf = (lane < 16) ? 0 : 8;     // 16-bit WMMA K-striping per lane group
  v8f acc = {0.f, 0.f, 0.f, 0.f, 0.f, 0.f, 0.f, 0.f};
  const __bf16* wbase = Wpack + (size_t)n * 32 + khalf;

#pragma unroll 4
  for (int kc = 0; kc < NKC; ++kc) {
    v8bf a_lo = *(const v8bf*)&Abuf[mrow][kc * 32 + khalf];
    v8bf a_hi = *(const v8bf*)&Abuf[mrow][kc * 32 + khalf + 16];
    const __bf16* wb = wbase + (size_t)kc * DMODEL * 32;
    v8bf b_lo = *(const v8bf*)wb;
    v8bf b_hi = *(const v8bf*)(wb + 16);
    v16bf A, B;
#pragma unroll
    for (int z = 0; z < 8; ++z) {
      A[z] = a_lo[z]; A[z + 8] = a_hi[z];
      B[z] = b_lo[z]; B[z + 8] = b_hi[z];
    }
    acc = __builtin_amdgcn_wmma_f32_16x16x32_bf16(false, A, false, B,
                                                  (short)0, acc, false, false);
  }

  // scatter C tile (+bias) to LDS: VGPR r, lane l -> M = r + 8*(l>=16), N = l&15
  const float bn = bias[n];
#pragma unroll
  for (int r = 0; r < 8; ++r) {
    int m = r + ((lane >= 16) ? 8 : 0);
    Etile[m][n] = acc[r] + bn;
  }
  __syncthreads();

  // LayerNorm over 128: each wave handles two rows
#pragma unroll
  for (int rr = 0; rr < 2; ++rr) {
    int row = wave * 2 + rr;
    float x0 = Etile[row][lane],      x1 = Etile[row][lane + 32];
    float x2 = Etile[row][lane + 64], x3 = Etile[row][lane + 96];
    float sum = x0 + x1 + x2 + x3;
    float ss  = x0 * x0 + x1 * x1 + x2 * x2 + x3 * x3;
#pragma unroll
    for (int off = 16; off > 0; off >>= 1) {
      sum += __shfl_xor(sum, off, 32);
      ss  += __shfl_xor(ss,  off, 32);
    }
    float mean = sum * (1.0f / 128.0f);
    float var  = ss * (1.0f / 128.0f) - mean * mean;
    float rstd = rsqrtf(var + 1e-5f);
    size_t e = (size_t)(ebase + row);
    for (int c = lane; c < DMODEL; c += 32) {
      float x = Etile[row][c];
      outE[e * DMODEL + c] = (x - mean) * rstd * g[c] + be[c];
    }
  }
}

// ---------------------------------------------------------------- launch
extern "C" void kernel_launch(void* const* d_in, const int* in_sizes, int n_in,
                              void* d_out, int out_size, void* d_ws, size_t ws_size,
                              hipStream_t stream) {
  const float* X       = (const float*)d_in[0];
  const int*   S       = (const int*)d_in[1];
  const float* BBD     = (const float*)d_in[2];
  const float* mask    = (const float*)d_in[3];
  const float* node_w  = (const float*)d_in[4];
  const float* node_b  = (const float*)d_in[5];
  const float* node_g  = (const float*)d_in[6];
  const float* node_be = (const float*)d_in[7];
  const float* edge_w  = (const float*)d_in[8];
  const float* edge_b  = (const float*)d_in[9];
  const float* edge_g  = (const float*)d_in[10];
  const float* edge_be = (const float*)d_in[11];

  float* out = (float*)d_out;
  char*  ws  = (char*)d_ws;
  float*  Xa    = (float*)(ws + WS_XA);
  int*    Eidx  = (int*)(ws + WS_EIDX);
  __bf16* Wpack = (__bf16*)(ws + WS_WPK);

  xa_kernel<<<(NNODES + 255) / 256, 256, 0, stream>>>(X, Xa);
  knn_kernel<<<NNODES, 256, 0, stream>>>(X, mask, Eidx, out + OUT_EIDX);
  pack_kernel<<<(NKC * DMODEL * 32 + 255) / 256, 256, 0, stream>>>(edge_w, Wpack);
  node_kernel<<<NNODES / 8, 256, 0, stream>>>(S, BBD, node_w, node_b, node_g, node_be,
                                              out + OUT_V);
  edge_kernel<<<NEDGES / 16, 256, 0, stream>>>(Xa, Eidx, Wpack, edge_b, edge_g, edge_be,
                                               out + OUT_E);
}
